// Attention_46634754900080
// MI455X (gfx1250) — matrix operand order repaired and checked by reading, not compiled, not run on hardware
//
#include <hip/hip_runtime.h>
#include <hip/hip_bf16.h>

// ---------------------------------------------------------------------------
// Attention forward for MI455X (gfx1250): bf16 WMMA GEMMs, async-to-LDS
// double-buffered staging, f32 softmax.
// Pipeline: cvt(x), transpose-cvt(W) -> QKV proj (V transposed) ->
//           QK^T+mask -> softmax -> PV.
// ---------------------------------------------------------------------------

typedef __attribute__((ext_vector_type(16))) __bf16 v16bf;
typedef __attribute__((ext_vector_type(8)))  __bf16 v8bf;
typedef __attribute__((ext_vector_type(4)))  __bf16 v4bf;
typedef __attribute__((ext_vector_type(8)))  float  v8f;

#define B_  8
#define S_  2048
#define D_  1024
#define U_  1024
#define BS_ (B_ * S_)
#define NEG_INF_ (-10000.0f)

// ---------------------------------------------------------------------------
// Async copy: one 16-byte chunk global -> LDS, tracked by ASYNCcnt.
// ---------------------------------------------------------------------------
__device__ __forceinline__ void async_copy_b128(const __bf16* lds_dst, const __bf16* gsrc) {
  unsigned lds_off = (unsigned)(size_t)lds_dst;           // low 32 bits = LDS addr
  unsigned long long ga = (unsigned long long)(size_t)gsrc;
  asm volatile("global_load_async_to_lds_b128 %0, %1, off"
               :: "v"(lds_off), "v"(ga) : "memory");
}
__device__ __forceinline__ void wait_async0() {
  asm volatile("s_wait_asynccnt 0x0" ::: "memory");
}

// Stage a 64-row x 32-col bf16 tile (rows contiguous, stride grs elements).
// 256 threads x 16B = 4 KB. One async b128 per thread.
__device__ __forceinline__ void stage_tile_async(__bf16* lds2048, const __bf16* gsrc,
                                                 int grs, int t) {
  int e = t * 8;
  int rl = e >> 5;        // 0..63
  int cl = e & 31;        // multiple of 8
  async_copy_b128(lds2048 + e, gsrc + (size_t)rl * grs + cl);
}

// ---------------------------------------------------------------------------
// f32 -> bf16 elementwise convert (n must be a multiple of 2048)
// ---------------------------------------------------------------------------
__global__ __launch_bounds__(256) void cvt_f32_bf16(const float* __restrict__ src,
                                                    __bf16* __restrict__ dst, int n) {
  int i = (blockIdx.x * 256 + threadIdx.x) * 8;
  if (i + 8 <= n) {
    float4 a0 = *(const float4*)(src + i);
    float4 a1 = *(const float4*)(src + i + 4);
    v8bf o;
    o[0] = (__bf16)a0.x; o[1] = (__bf16)a0.y; o[2] = (__bf16)a0.z; o[3] = (__bf16)a0.w;
    o[4] = (__bf16)a1.x; o[5] = (__bf16)a1.y; o[6] = (__bf16)a1.z; o[7] = (__bf16)a1.w;
    *(v8bf*)(dst + i) = o;
  }
}

// ---------------------------------------------------------------------------
// Transpose-convert one weight matrix: src f32 [D][U] -> dst bf16 [U][D].
// 32x32 tiles through LDS so both global sides are contiguous.
// ---------------------------------------------------------------------------
__global__ __launch_bounds__(256) void cvt_w_transpose(const float* __restrict__ src,
                                                       __bf16* __restrict__ dst) {
  __shared__ float tile[32][33];
  const int t = threadIdx.x;
  const int c0 = blockIdx.x * 32;   // U tile
  const int r0 = blockIdx.y * 32;   // D tile
  {
    int e = t * 4, rl = e >> 5, cl = e & 31;
    float4 v = *(const float4*)(src + (size_t)(r0 + rl) * U_ + c0 + cl);
    tile[rl][cl + 0] = v.x; tile[rl][cl + 1] = v.y;
    tile[rl][cl + 2] = v.z; tile[rl][cl + 3] = v.w;
  }
  __syncthreads();
  {
    int e = t * 4, ul = e >> 5, dl = e & 31;
    v4bf o;
    o[0] = (__bf16)tile[dl + 0][ul];
    o[1] = (__bf16)tile[dl + 1][ul];
    o[2] = (__bf16)tile[dl + 2][ul];
    o[3] = (__bf16)tile[dl + 3][ul];
    *(v4bf*)(dst + (size_t)(c0 + ul) * D_ + r0 + dl) = o;
  }
}

// ---------------------------------------------------------------------------
// WMMA fragment loaders (layouts per CDNA5 ISA).
// A: 16x32 bf16 from row-major matrix. Lanes 0-15: M=lane, elems 0-7 K=k0..+7,
// elems 8-15 K=k0+16..+23; lanes 16-31 shift K by +8.
// ---------------------------------------------------------------------------
__device__ __forceinline__ v16bf load_a16(const __bf16* __restrict__ base, int ld,
                                          int row_base, int k0, int lane) {
  int m  = lane & 15;
  int kh = (lane >> 4) * 8;
  const __bf16* p = base + (size_t)(row_base + m) * ld + k0 + kh;
  union { v16bf v; v8bf h[2]; } u;
  u.h[0] = *(const v8bf*)(p);
  u.h[1] = *(const v8bf*)(p + 16);
  return u.v;
}

// B: from LDS tile laid out [n(64)][k(32)] row-major: lane holds column
// N = nn*16 + (lane&15), elems 0-7 K=(lane>>4)*8..+7, elems 8-15 K=16+(lane>>4)*8..+7.
__device__ __forceinline__ v16bf load_b16_lds(const __bf16* lds, int nn, int lane) {
  const __bf16* p = lds + (size_t)(nn * 16 + (lane & 15)) * 32 + (lane >> 4) * 8;
  union { v16bf v; v8bf h[2]; } u;
  u.h[0] = *(const v8bf*)(p);
  u.h[1] = *(const v8bf*)(p + 16);
  return u.v;
}

// One 32-deep K step for a 32(M) x 64(N) wave tile: 2 A-frags x 4 B-frags,
// 8 independent WMMAs (hides WMMA->WMMA hazard latency).
__device__ __forceinline__ void gemm_step(const __bf16* __restrict__ Abase, int lda,
                                          int row0, int kstep, const __bf16* ldsTile,
                                          int lane, v8f acc[8]) {
  v16bf a0 = load_a16(Abase, lda, row0, kstep, lane);
  v16bf a1 = load_a16(Abase, lda, row0 + 16, kstep, lane);
#pragma unroll
  for (int nn = 0; nn < 4; ++nn) {
    v16bf bf = load_b16_lds(ldsTile, nn, lane);
    acc[nn]     = __builtin_amdgcn_wmma_f32_16x16x32_bf16(false, a0, false, bf,
                                                          (short)0, acc[nn], false, false);
    acc[4 + nn] = __builtin_amdgcn_wmma_f32_16x16x32_bf16(false, a1, false, bf,
                                                          (short)0, acc[4 + nn], false, false);
  }
}

#define ZACC8 { {0.f,0.f,0.f,0.f,0.f,0.f,0.f,0.f}, {0.f,0.f,0.f,0.f,0.f,0.f,0.f,0.f}, \
                {0.f,0.f,0.f,0.f,0.f,0.f,0.f,0.f}, {0.f,0.f,0.f,0.f,0.f,0.f,0.f,0.f}, \
                {0.f,0.f,0.f,0.f,0.f,0.f,0.f,0.f}, {0.f,0.f,0.f,0.f,0.f,0.f,0.f,0.f}, \
                {0.f,0.f,0.f,0.f,0.f,0.f,0.f,0.f}, {0.f,0.f,0.f,0.f,0.f,0.f,0.f,0.f} }

// ---------------------------------------------------------------------------
// QKV projection: out = xb[BS,D] @ W + bias, W given transposed bf16 [U][D].
// Block: 8 waves; wave tile 32x64; block tile 256x64. Async double-buffered
// B-tile staging ([n][k] rows contiguous since W is pre-transposed).
// Q,K stored row-major bf16 [BS,U]; V stored transposed bf16 [B][U][S].
// ---------------------------------------------------------------------------
__global__ __launch_bounds__(256) void qkv_proj(
    const __bf16* __restrict__ xb, const __bf16* __restrict__ Wt,
    const float* __restrict__ bq, const float* __restrict__ bk,
    const float* __restrict__ bv,
    __bf16* __restrict__ Qb, __bf16* __restrict__ Kb, __bf16* __restrict__ Vt) {
  __shared__ __attribute__((aligned(32))) __bf16 tiles[2][64 * 32];
  const int t = threadIdx.x, lane = t & 31, wave = t >> 5;
  const int z = blockIdx.z;
  const int nbase = blockIdx.x * 64;
  const int mbase = blockIdx.y * 256 + wave * 32;
  const __bf16* W = Wt + (size_t)z * U_ * D_ + (size_t)nbase * D_;  // rows = n, ld = D_
  const float* bias = (z == 0) ? bq : (z == 1) ? bk : bv;

  v8f acc[8] = ZACC8;

  stage_tile_async(tiles[0], W, D_, t);
  wait_async0();
  __syncthreads();
  const int KSTEPS = D_ / 32;
  for (int kb = 0; kb < KSTEPS; ++kb) {
    int cur = kb & 1;
    if (kb + 1 < KSTEPS) stage_tile_async(tiles[cur ^ 1], W + (kb + 1) * 32, D_, t);
    gemm_step(xb, D_, mbase, kb * 32, tiles[cur], lane, acc);
    wait_async0();
    __syncthreads();
  }

  const int mo = (lane >> 4) * 8, nlo = lane & 15;
#pragma unroll
  for (int mi = 0; mi < 2; ++mi) {
#pragma unroll
    for (int nn = 0; nn < 4; ++nn) {
      int col = nbase + nn * 16 + nlo;
      float bval = bias[col];
#pragma unroll
      for (int r = 0; r < 8; ++r) {
        int row = mbase + mi * 16 + mo + r;
        __bf16 h = (__bf16)(acc[mi * 4 + nn][r] + bval);
        if (z == 0)
          Qb[(size_t)row * U_ + col] = h;
        else if (z == 1)
          Kb[(size_t)row * U_ + col] = h;
        else {
          int bi = row >> 11;          // row / S_
          int si = row & (S_ - 1);
          Vt[((size_t)bi * U_ + col) * S_ + si] = h;
        }
      }
    }
  }
}

// ---------------------------------------------------------------------------
// scores[b,q,k] = (Q[b,q,:].K[b,k,:]) / sqrt(U) + (1 - mq*mk)*NEG_INF.
// Row-major K matches the B-fragment shape (rows = keys, 32 u per tile).
// ---------------------------------------------------------------------------
__global__ __launch_bounds__(256) void qk_scores(
    const __bf16* __restrict__ Qb, const __bf16* __restrict__ Kb,
    const int* __restrict__ mask, float* __restrict__ scores) {
  __shared__ __attribute__((aligned(32))) __bf16 tiles[2][64 * 32];
  const int t = threadIdx.x, lane = t & 31, wave = t >> 5;
  const int b = blockIdx.z;
  const int keybase = blockIdx.x * 64;
  const int qbase = blockIdx.y * 256 + wave * 32;
  const __bf16* Kt = Kb + (size_t)b * S_ * U_ + (size_t)keybase * U_;  // rows = keys

  v8f acc[8] = ZACC8;

  stage_tile_async(tiles[0], Kt, U_, t);
  wait_async0();
  __syncthreads();
  const int KSTEPS = U_ / 32;
  for (int ub = 0; ub < KSTEPS; ++ub) {
    int cur = ub & 1;
    if (ub + 1 < KSTEPS) stage_tile_async(tiles[cur ^ 1], Kt + (ub + 1) * 32, U_, t);
    gemm_step(Qb, U_, b * S_ + qbase, ub * 32, tiles[cur], lane, acc);
    wait_async0();
    __syncthreads();
  }

  const float scale = 0.03125f;  // 1/sqrt(1024)
  const int mo = (lane >> 4) * 8, nlo = lane & 15;
#pragma unroll
  for (int mi = 0; mi < 2; ++mi) {
#pragma unroll
    for (int nn = 0; nn < 4; ++nn) {
      int key = keybase + nn * 16 + nlo;
      float mk = (float)mask[b * S_ + key];
#pragma unroll
      for (int r = 0; r < 8; ++r) {
        int q = qbase + mi * 16 + mo + r;
        float mq = (float)mask[b * S_ + q];
        scores[((size_t)b * S_ + q) * S_ + key] =
            acc[mi * 4 + nn][r] * scale + (1.0f - mq * mk) * NEG_INF_;
      }
    }
  }
}

// ---------------------------------------------------------------------------
// Row softmax: f32 in, bf16 probs out. One 256-thread block per row (S=2048).
// ---------------------------------------------------------------------------
__global__ __launch_bounds__(256) void softmax_rows(const float* __restrict__ scores,
                                                    __bf16* __restrict__ P) {
  __shared__ float red[256];
  const int row = blockIdx.x, t = threadIdx.x;
  const float* s = scores + (size_t)row * S_;

  float mx = -3.0e38f;
  for (int i = t; i < S_; i += 256) mx = fmaxf(mx, s[i]);
  red[t] = mx;
  __syncthreads();
  for (int o = 128; o > 0; o >>= 1) {
    if (t < o) red[t] = fmaxf(red[t], red[t + o]);
    __syncthreads();
  }
  mx = red[0];
  __syncthreads();

  float sum = 0.0f;
  for (int i = t; i < S_; i += 256) sum += __expf(s[i] - mx);
  red[t] = sum;
  __syncthreads();
  for (int o = 128; o > 0; o >>= 1) {
    if (t < o) red[t] += red[t + o];
    __syncthreads();
  }
  float inv = 1.0f / red[0];

  __bf16* p = P + (size_t)row * S_;
  for (int i = t; i < S_; i += 256) p[i] = (__bf16)(__expf(s[i] - mx) * inv);
}

// ---------------------------------------------------------------------------
// context[b] = P[b] (SxS bf16) @ V[b].  V pre-transposed [B][U][S] so the
// B-tile rows (fixed u, 32 contiguous keys) stage with async b128 copies.
// ---------------------------------------------------------------------------
__global__ __launch_bounds__(256) void pv_context(const __bf16* __restrict__ P,
                                                  const __bf16* __restrict__ Vt,
                                                  float* __restrict__ out) {
  __shared__ __attribute__((aligned(32))) __bf16 tiles[2][64 * 32];
  const int t = threadIdx.x, lane = t & 31, wave = t >> 5;
  const int b = blockIdx.z;
  const int ubase = blockIdx.x * 64;
  const int qbase = blockIdx.y * 256 + wave * 32;
  const __bf16* Pb = P + (size_t)b * S_ * S_;
  const __bf16* Vb = Vt + (size_t)b * U_ * S_ + (size_t)ubase * S_;  // rows = u

  v8f acc[8] = ZACC8;

  stage_tile_async(tiles[0], Vb, S_, t);
  wait_async0();
  __syncthreads();
  const int KSTEPS = S_ / 32;
  for (int kb = 0; kb < KSTEPS; ++kb) {
    int cur = kb & 1;
    if (kb + 1 < KSTEPS) stage_tile_async(tiles[cur ^ 1], Vb + (kb + 1) * 32, S_, t);
    gemm_step(Pb, S_, qbase, kb * 32, tiles[cur], lane, acc);
    wait_async0();
    __syncthreads();
  }

  const int mo = (lane >> 4) * 8, nlo = lane & 15;
#pragma unroll
  for (int mi = 0; mi < 2; ++mi) {
#pragma unroll
    for (int nn = 0; nn < 4; ++nn) {
      int u = ubase + nn * 16 + nlo;
#pragma unroll
      for (int r = 0; r < 8; ++r) {
        int q = qbase + mi * 16 + mo + r;
        out[((size_t)b * S_ + q) * U_ + u] = acc[mi * 4 + nn][r];
      }
    }
  }
}

// ---------------------------------------------------------------------------
extern "C" void kernel_launch(void* const* d_in, const int* in_sizes, int n_in,
                              void* d_out, int out_size, void* d_ws, size_t ws_size,
                              hipStream_t stream) {
  const float* x    = (const float*)d_in[0];
  const int*   mask = (const int*)d_in[1];
  const float* Wq   = (const float*)d_in[2];
  const float* bq   = (const float*)d_in[3];
  const float* Wk   = (const float*)d_in[4];
  const float* bk   = (const float*)d_in[5];
  const float* Wv   = (const float*)d_in[6];
  const float* bv   = (const float*)d_in[7];
  float* out = (float*)d_out;

  char* ws = (char*)d_ws;
  size_t off = 0;
  auto alloc = [&](size_t bytes) {
    void* p = ws + off;
    off = (off + bytes + 255) & ~(size_t)255;
    return p;
  };

  __bf16* xb     = (__bf16*)alloc((size_t)BS_ * D_ * 2);       //  32 MB
  __bf16* Wt     = (__bf16*)alloc((size_t)3 * U_ * D_ * 2);    //   6 MB  [3][U][D]
  __bf16* Qb     = (__bf16*)alloc((size_t)BS_ * U_ * 2);       //  32 MB
  __bf16* Kb     = (__bf16*)alloc((size_t)BS_ * U_ * 2);       //  32 MB
  __bf16* Vt     = (__bf16*)alloc((size_t)BS_ * U_ * 2);       //  32 MB  [B][U][S]
  float*  scores = (float*)alloc((size_t)B_ * S_ * S_ * 4);    // 128 MB
  __bf16* Pb     = (__bf16*)alloc((size_t)B_ * S_ * S_ * 2);   //  64 MB

  // 1) precision conversion (+ weight transpose so GEMM B-tiles are contiguous)
  cvt_f32_bf16<<<(BS_ * D_) / 2048, 256, 0, stream>>>(x, xb, BS_ * D_);
  cvt_w_transpose<<<dim3(U_ / 32, D_ / 32), 256, 0, stream>>>(Wq, Wt);
  cvt_w_transpose<<<dim3(U_ / 32, D_ / 32), 256, 0, stream>>>(Wk, Wt + (size_t)U_ * D_);
  cvt_w_transpose<<<dim3(U_ / 32, D_ / 32), 256, 0, stream>>>(Wv, Wt + (size_t)2 * U_ * D_);

  // 2) Q/K/V projections (z = 0/1/2)
  qkv_proj<<<dim3(U_ / 64, BS_ / 256, 3), 256, 0, stream>>>(xb, Wt, bq, bk, bv, Qb, Kb, Vt);

  // 3) masked scaled scores
  qk_scores<<<dim3(S_ / 64, S_ / 256, B_), 256, 0, stream>>>(Qb, Kb, mask, scores);

  // 4) softmax -> bf16 probs
  softmax_rows<<<B_ * S_, 256, 0, stream>>>(scores, Pb);

  // 5) context = P @ V
  pv_context<<<dim3(U_ / 64, S_ / 256, B_), 256, 0, stream>>>(Pb, Vt, out);
}
